// bahdanau_attention_16733192585556
// MI455X (gfx1250) — matrix operand order repaired and checked by reading, not compiled, not run on hardware
//
#include <hip/hip_runtime.h>
#include <hip/hip_bf16.h>
#include <math.h>

typedef __attribute__((ext_vector_type(16))) __bf16 v16bf;
typedef __attribute__((ext_vector_type(8)))  float  v8f;

#define Bsz 32
#define Ssz 4096
#define Hsz 512
#define Esz 512
#define LDS_STRIDE 520   // 512 + 8 bf16 pad -> row byte stride 1040 (16B multiple, staggers banks)

__device__ __forceinline__ float fast_tanh(float x) {
    // tanh(x) = 1 - 2/(exp(2x)+1); exp->v_exp_f32, rcp->v_rcp_f32 (TRANS, co-executes with WMMA)
    float e = __expf(2.0f * x);
    return 1.0f - 2.0f * __builtin_amdgcn_rcpf(e + 1.0f);
}

// ---------------- We -> bf16 ----------------
__global__ void convert_we_kernel(const float* __restrict__ We, __bf16* __restrict__ Web) {
    int i = blockIdx.x * blockDim.x + threadIdx.x;
    int stride = gridDim.x * blockDim.x;
    for (; i < Hsz * Hsz; i += stride) Web[i] = (__bf16)We[i];
}

// ---------------- gd = h @ Wd^T + bd ----------------
__global__ void gd_kernel(const float* __restrict__ h, const float* __restrict__ Wd,
                          const float* __restrict__ bd, float* __restrict__ gd) {
    int b = blockIdx.x;
    int t = threadIdx.x;
    __shared__ float hs[Hsz];
    hs[t]       = h[b * Hsz + t];
    hs[t + 256] = h[b * Hsz + t + 256];
    __syncthreads();
    for (int q = 0; q < 2; q++) {
        int k = t + q * 256;
        const float* wr = Wd + (size_t)k * Hsz;
        float acc = bd[k];
        for (int i = 0; i < Hsz; i++) acc += hs[i] * wr[i];
        gd[b * Hsz + k] = acc;
    }
}

// ---------------- scores: WMMA bf16 GEMM fused with tanh / Wv dot ----------------
// grid: 1024 blocks (b * 32 s-tiles), 256 threads = 8 waves
// LDS: A tile 128x520 bf16 + double-buffered B chunk 2 x 16x520 bf16 + bias/wv floats
__global__ __launch_bounds__(256) void scores_kernel(
        const float* __restrict__ ctx, const __bf16* __restrict__ Web,
        const float* __restrict__ be, const float* __restrict__ gd,
        const float* __restrict__ Wv, const float* __restrict__ bvp,
        float* __restrict__ scores) {
    extern __shared__ __bf16 smem[];
    __bf16* As    = smem;                              // [128][LDS_STRIDE]
    __bf16* B0    = smem + 128 * LDS_STRIDE;           // [16][LDS_STRIDE]
    __bf16* B1    = B0 + 16 * LDS_STRIDE;              // [16][LDS_STRIDE]
    float*  biasS = (float*)(B1 + 16 * LDS_STRIDE);    // [512]
    float*  wvS   = biasS + Hsz;                       // [512]

    const int b  = blockIdx.x >> 5;
    const int st = blockIdx.x & 31;
    const int s0 = st * 128;
    const int t  = threadIdx.x;

    // ---- stage 128x512 context tile into LDS as bf16 (8 floats per chunk) ----
    const size_t ctx_base = ((size_t)b * Ssz + s0) * Hsz;
    for (int chunk = t; chunk < 8192; chunk += 256) {
        int row = chunk >> 6;                // 64 chunks of 8 floats per row
        int c8  = chunk & 63;
        const float4* src = (const float4*)(ctx + ctx_base + (size_t)row * Hsz + c8 * 8);
        float4 f0 = src[0];
        float4 f1 = src[1];
        union { uint4 q; __bf16 v[8]; } pk;
        pk.v[0] = (__bf16)f0.x; pk.v[1] = (__bf16)f0.y;
        pk.v[2] = (__bf16)f0.z; pk.v[3] = (__bf16)f0.w;
        pk.v[4] = (__bf16)f1.x; pk.v[5] = (__bf16)f1.y;
        pk.v[6] = (__bf16)f1.z; pk.v[7] = (__bf16)f1.w;
        *(uint4*)(As + (size_t)row * LDS_STRIDE + c8 * 8) = pk.q;
    }

    // ---- stage B chunk for n0=0 into B0 (16 rows x 512 bf16 = 1024 uint4) ----
    {
        const uint4* src = (const uint4*)(Web);
        for (int i = t; i < 1024; i += 256) {
            int row = i >> 6;                // 64 uint4 per row
            int q   = i & 63;
            *(uint4*)(B0 + (size_t)row * LDS_STRIDE + q * 8) = src[(size_t)row * 64 + q];
        }
    }
    // ---- stage bias (be + gd[b]) and Wv into LDS ----
    for (int q = 0; q < 2; q++) {
        int n = t + q * 256;
        biasS[n] = be[n] + gd[b * Hsz + n];
        wvS[n]   = Wv[n];
    }
    __syncthreads();

    const int wave = t >> 5;
    const int lane = t & 31;
    const int m    = lane & 15;              // A: row M ; B/D: col N
    const int half = lane >> 4;
    const int w16  = wave * 16;              // this wave's 16 rows within the tile

    // ---- hoist all 16 A fragments (K=0..511) into registers: invariant across n0 ----
    const __bf16* arow = As + (size_t)(w16 + m) * LDS_STRIDE;
    v16bf afrag[16];
#pragma unroll
    for (int ki = 0; ki < 16; ki++) {
        union { uint4 q[2]; v16bf v; } av;
        // A 16x32 bf16 layout: lanes<16 K={0..7,16..23}, lanes>=16 K={8..15,24..31}
        av.q[0] = *(const uint4*)(arow + ki * 32 + half * 8);
        av.q[1] = *(const uint4*)(arow + ki * 32 + 16 + half * 8);
        afrag[ki] = av.v;
    }

    float srow[8];
#pragma unroll
    for (int r = 0; r < 8; r++) srow[r] = 0.f;

    for (int n0 = 0; n0 < Hsz; n0 += 16) {
        const __bf16* Bcur = ((n0 >> 4) & 1) ? B1 : B0;
        __bf16*       Bnxt = ((n0 >> 4) & 1) ? B0 : B1;

        // prefetch next 16 We rows into the other buffer (overlaps with compute below)
        if (n0 + 16 < Hsz) {
            const uint4* src = (const uint4*)(Web + (size_t)(n0 + 16) * Hsz);
            for (int i = t; i < 1024; i += 256) {
                int row = i >> 6;
                int q   = i & 63;
                *(uint4*)(Bnxt + (size_t)row * LDS_STRIDE + q * 8) = src[(size_t)row * 64 + q];
            }
        }

        const int n = n0 + m;                // output column handled by this lane
        const float bias = biasS[n];
        const float wv   = wvS[n];
        // B fragment: B[K=h, N=n] = We[n, h]; lane holds K={half*8..+7, 16+half*8..+7}
        const __bf16* bp = Bcur + (size_t)m * LDS_STRIDE + half * 8;

        v8f acc = {0.f, 0.f, 0.f, 0.f, 0.f, 0.f, 0.f, 0.f};
#pragma unroll
        for (int ki = 0; ki < 16; ki++) {
            union { uint4 q[2]; v16bf v; } bv_;
            // B 32x16 bf16 layout: lanes<16 K={0..7,16..23}, lanes>=16 K={8..15,24..31}
            bv_.q[0] = *(const uint4*)(bp + ki * 32);
            bv_.q[1] = *(const uint4*)(bp + ki * 32 + 16);
            acc = __builtin_amdgcn_wmma_f32_16x16x32_bf16(
                false, afrag[ki], false, bv_.v, (short)0, acc, false, false);
        }
        // D layout: lane N = lane&15, VGPR r holds M = r + 8*half
#pragma unroll
        for (int r = 0; r < 8; r++) srow[r] += wv * fast_tanh(acc[r] + bias);

        __syncthreads();   // next-buffer writes visible; current-buffer reads done
    }

    // reduce over the 16 N-lanes within each half (xor<16 stays in-half)
#pragma unroll
    for (int off = 1; off < 16; off <<= 1) {
#pragma unroll
        for (int r = 0; r < 8; r++) srow[r] += __shfl_xor(srow[r], off, 32);
    }

    if (m == 0) {
        const float bv0 = bvp[0];
        const int sbase = b * Ssz + s0 + w16 + half * 8;
#pragma unroll
        for (int r = 0; r < 8; r++) scores[sbase + r] = srow[r] + bv0;
    }
}

// ---------------- softmax over S per batch ----------------
__global__ void softmax_kernel(const float* __restrict__ scores, float* __restrict__ weights) {
    int b = blockIdx.x;
    int t = threadIdx.x;
    __shared__ float red[256];
    const float* s = scores + (size_t)b * Ssz;
    float v[16];
    float mx = -INFINITY;
#pragma unroll
    for (int i = 0; i < 16; i++) { v[i] = s[t + i * 256]; mx = fmaxf(mx, v[i]); }
    red[t] = mx; __syncthreads();
    for (int o = 128; o > 0; o >>= 1) { if (t < o) red[t] = fmaxf(red[t], red[t + o]); __syncthreads(); }
    mx = red[0]; __syncthreads();
    float sum = 0.f;
#pragma unroll
    for (int i = 0; i < 16; i++) { v[i] = __expf(v[i] - mx); sum += v[i]; }
    red[t] = sum; __syncthreads();
    for (int o = 128; o > 0; o >>= 1) { if (t < o) red[t] += red[t + o]; __syncthreads(); }
    float inv = 1.0f / red[0];
    float* w = weights + (size_t)b * Ssz;
#pragma unroll
    for (int i = 0; i < 16; i++) w[t + i * 256] = v[i] * inv;
}

// ---------------- c_t partials: grid (32 b, 16 s-chunks) ----------------
__global__ void ctx_part_kernel(const float* __restrict__ ctx, const float* __restrict__ weights,
                                float* __restrict__ c_part) {
    int b = blockIdx.x, ch = blockIdx.y, t = threadIdx.x;
    const float*  w  = weights + (size_t)b * Ssz + ch * 256;
    const float2* cb = (const float2*)(ctx + ((size_t)b * Ssz + (size_t)ch * 256) * Hsz);
    float2 acc = {0.f, 0.f};
    for (int s = 0; s < 256; s++) {
        float ws_ = w[s];
        float2 c  = cb[(size_t)s * 256 + t];
        acc.x += ws_ * c.x;
        acc.y += ws_ * c.y;
    }
    ((float2*)(c_part + ((size_t)b * 16 + ch) * Hsz))[t] = acc;
}

__global__ void ctx_reduce_kernel(const float* __restrict__ c_part, float* __restrict__ c_t) {
    int b = blockIdx.x, t = threadIdx.x;
    for (int q = 0; q < 2; q++) {
        int hcol = t + q * 256;
        float a = 0.f;
        for (int ch = 0; ch < 16; ch++) a += c_part[((size_t)b * 16 + ch) * Hsz + hcol];
        c_t[b * Hsz + hcol] = a;
    }
}

// ---------------- r_t = [c_t,h,x] @ Wr^T + br ; maxout pairs ----------------
__global__ void out_kernel(const float* __restrict__ c_t, const float* __restrict__ h,
                           const float* __restrict__ x, const float* __restrict__ Wr,
                           const float* __restrict__ br, float* __restrict__ out) {
    int b = blockIdx.x, t = threadIdx.x;
    __shared__ float cat[3 * Hsz];
    for (int q = 0; q < 2; q++) {
        int i = q * 256 + t;
        cat[i]            = c_t[b * Hsz + i];
        cat[Hsz + i]      = h[b * Hsz + i];
        cat[2 * Hsz + i]  = x[b * Esz + i];
    }
    __syncthreads();
    for (int q = 0; q < 2; q++) {
        int p  = t + q * 256;                 // output index 0..511
        int j0 = 2 * p;
        const float* w0 = Wr + (size_t)j0 * (3 * Hsz);
        const float* w1 = w0 + 3 * Hsz;
        float a0 = br[j0], a1 = br[j0 + 1];
        for (int i = 0; i < 3 * Hsz; i++) {
            float c = cat[i];
            a0 += c * w0[i];
            a1 += c * w1[i];
        }
        out[b * Hsz + p] = fmaxf(a0, a1);
    }
}

extern "C" void kernel_launch(void* const* d_in, const int* in_sizes, int n_in,
                              void* d_out, int out_size, void* d_ws, size_t ws_size,
                              hipStream_t stream) {
    (void)in_sizes; (void)n_in; (void)out_size; (void)ws_size;
    const float* ctx = (const float*)d_in[0];
    const float* h   = (const float*)d_in[1];
    const float* x   = (const float*)d_in[2];
    const float* We  = (const float*)d_in[3];
    const float* be  = (const float*)d_in[4];
    const float* Wd  = (const float*)d_in[5];
    const float* bd  = (const float*)d_in[6];
    const float* Wv  = (const float*)d_in[7];
    const float* bv  = (const float*)d_in[8];
    const float* Wr  = (const float*)d_in[9];
    const float* br  = (const float*)d_in[10];

    float* out     = (float*)d_out;              // [32,512] maxout output
    float* weights = out + Bsz * Hsz;            // [32,4096] softmax weights (2nd output)

    char*   ws     = (char*)d_ws;
    __bf16* Web    = (__bf16*)ws;                          // 512*512*2      = 524288 B
    float*  gd     = (float*)(ws + 524288);                // 32*512*4       =  65536 B
    float*  scores = (float*)(ws + 524288 + 65536);        // 32*4096*4      = 524288 B
    float*  c_t    = (float*)(ws + 524288 + 65536 + 524288);           // 32*512*4  = 65536 B
    float*  c_part = (float*)(ws + 524288 + 65536 + 524288 + 65536);   // 32*16*512*4 = 1 MB

    convert_we_kernel<<<256, 256, 0, stream>>>(We, Web);
    gd_kernel<<<Bsz, 256, 0, stream>>>(h, Wd, bd, gd);

    const size_t lds_bytes = (size_t)(128 + 32) * LDS_STRIDE * sizeof(__bf16)
                           + (size_t)2 * Hsz * sizeof(float);           // 170496 B
    scores_kernel<<<Bsz * (Ssz / 128), 256, lds_bytes, stream>>>(ctx, Web, be, gd, Wv, bv, scores);

    softmax_kernel<<<Bsz, 256, 0, stream>>>(scores, weights);
    ctx_part_kernel<<<dim3(Bsz, 16), 256, 0, stream>>>(ctx, weights, c_part);
    ctx_reduce_kernel<<<Bsz, 256, 0, stream>>>(c_part, c_t);
    out_kernel<<<Bsz, 256, 0, stream>>>(c_t, h, x, Wr, br, out);
}
